// ResModel_18176301597580
// MI455X (gfx1250) — matrix operand order repaired and checked by reading, not compiled, not run on hardware
//
#include <hip/hip_runtime.h>
#include <hip/hip_bf16.h>

// ---------------- problem constants (from reference) ----------------
#define N_NODES 80000
#define NCFG    8000
#define GG      16
#define CC      5
#define OPF     140
#define OPE     32
#define CFGF    18
#define HH      32
#define NFD     (OPF + OPE)      // 172
#define DIN     (CFGF + NFD)     // 190
#define DIN_PAD 192              // 6 k-tiles of 32
#define DGC     (CFGF + HH)      // 50
#define DGC_PAD 64               // 2 k-tiles of 32
#define ROWS    (N_NODES * CC)   // 400000 (divisible by 16)
#define NTILES  (ROWS / 16)      // 25000 (divisible by 8)
#define ALPHA_LR 0.2f

typedef __bf16 bf16_t;
typedef __attribute__((ext_vector_type(16))) __bf16 v16bf;
typedef __attribute__((ext_vector_type(8)))  __bf16 v8bf;
typedef __attribute__((ext_vector_type(8)))  float  v8f;

__device__ __forceinline__ float lrelu(float x) { return x >= 0.f ? x : ALPHA_LR * x; }

// A fragment (16x32 bf16, M x K): lane m=lane&15; kb = 0 or 8 by lane half;
// elements 0..7 = K kb..kb+7, elements 8..15 = K kb+16..kb+23  (ISA 7.12.2)
__device__ __forceinline__ v16bf load_a_tile(const bf16_t* base, int ldk, int lane) {
  int m  = lane & 15;
  int kb = (lane >> 4) << 3;
  const bf16_t* p = base + m * ldk + kb;
  v8bf lo = *(const v8bf*)(p);
  v8bf hi = *(const v8bf*)(p + 16);
  v16bf a;
#pragma unroll
  for (int i = 0; i < 8; ++i) { a[i] = lo[i]; a[i + 8] = hi[i]; }
  return a;
}

// B fragment (32x16 bf16, K x N) from pre-packed W^T [32][ldk]:
// lane n=n0+(lane&15); kb = k0 + (lane half ? 8 : 0); elements 0..7 = K kb..kb+7, 8..15 = K kb+16..kb+23
__device__ __forceinline__ v16bf load_b_tile(const bf16_t* wt, int ldk, int n0, int k0, int lane) {
  int n  = n0 + (lane & 15);
  int kb = k0 + ((lane >> 4) << 3);
  const bf16_t* p = wt + n * ldk + kb;
  v8bf lo = *(const v8bf*)(p);
  v8bf hi = *(const v8bf*)(p + 16);
  v16bf b;
#pragma unroll
  for (int i = 0; i < 8; ++i) { b[i] = lo[i]; b[i + 8] = hi[i]; }
  return b;
}

#define WMMA_BF16(a, b, c) \
  __builtin_amdgcn_wmma_f32_16x16x32_bf16(false, (a), false, (b), (short)0, (c), false, false)

// ---------------- small setup kernels ----------------

// pack W [K x 32] f32 -> W^T [32 x Kpad] bf16 (zero-padded K)
__global__ void pack_wT_kernel(const float* __restrict__ w, bf16_t* __restrict__ wt,
                               int K, int Kpad) {
  int tid = blockIdx.x * blockDim.x + threadIdx.x;
  if (tid >= HH * Kpad) return;
  int n = tid / Kpad, k = tid - n * Kpad;
  wt[tid] = (bf16_t)((k < K) ? w[k * HH + n] : 0.f);
}

// nf = concat(op_feats, emb[op_ids]) -> bf16 [N x 172]
__global__ void build_nf_kernel(const float* __restrict__ opf, const float* __restrict__ emb,
                                const int* __restrict__ op_ids, bf16_t* __restrict__ nfb) {
  int tid = blockIdx.x * blockDim.x + threadIdx.x;
  if (tid >= N_NODES * NFD) return;
  int node = tid / NFD, k = tid - node * NFD;
  float v = (k < OPF) ? opf[node * OPF + k] : emb[op_ids[node] * OPE + (k - OPF)];
  nfb[tid] = (bf16_t)v;
}

// cfg[node] += 100 * config_feats[src]  (segment_sum scatter)
__global__ void cfg_scatter_kernel(const float* __restrict__ cf, const int* __restrict__ csrc,
                                   const int* __restrict__ cdst, float* __restrict__ cfg) {
  int tid = blockIdx.x * blockDim.x + threadIdx.x;
  if (tid >= NCFG * CC) return;
  int j = tid / CC, c = tid - j * CC;
  int i = csrc[j], d = cdst[j];
  const float* s = cf + ((long)i * CC + c) * CFGF;
  float* o = cfg + ((long)d * CC + c) * CFGF;
#pragma unroll
  for (int k = 0; k < CFGF; ++k) atomicAdd(&o[k], 100.f * s[k]);
}

// degree over dst_all = [dst, src]  (self-loop +1 folded into inv kernel)
__global__ void deg_kernel(const int* __restrict__ src, const int* __restrict__ dst,
                           int E, float* __restrict__ deg) {
  int tid = blockIdx.x * blockDim.x + threadIdx.x;
  if (tid >= 2 * E) return;
  int d = (tid < E) ? dst[tid] : src[tid - E];
  atomicAdd(&deg[d], 1.f);
}

__global__ void inv_kernel(const float* __restrict__ deg, float* __restrict__ inv) {
  int tid = blockIdx.x * blockDim.x + threadIdx.x;
  if (tid >= N_NODES) return;
  inv[tid] = rsqrtf(deg[tid] + 1.0f);  // +1 = self loop
}

// ---------------- WMMA MLP kernels ----------------
// x = lrelu( lrelu([cfg, nf] @ W1 + b1) @ W2 + b2 )   K=190 (pad 192)
__global__ __launch_bounds__(256)
void mlp_pre_kernel(const bf16_t* __restrict__ nfb, const float* __restrict__ cfg,
                    const bf16_t* __restrict__ w1t, const float* __restrict__ b1,
                    const bf16_t* __restrict__ w2t, const float* __restrict__ b2,
                    float* __restrict__ xout) {
  __shared__ __align__(16) bf16_t sA[8 * 16 * DIN_PAD];
  __shared__ __align__(16) bf16_t sH[8 * 16 * HH];
  int wid = threadIdx.x >> 5, lane = threadIdx.x & 31;
  int tile = blockIdx.x * 8 + wid;
  bf16_t* at = sA + wid * 16 * DIN_PAD;
  bf16_t* ht = sH + wid * 16 * HH;
  int rbase = tile * 16;

  for (int j = lane; j < 16 * DIN_PAD; j += 32) {
    int row = j / DIN_PAD, k = j - row * DIN_PAD;
    int gr = rbase + row;
    int node = gr / CC, cc = gr - node * CC;
    bf16_t v;
    if (k < CFGF)      v = (bf16_t)cfg[((long)node * CC + cc) * CFGF + k];
    else if (k < DIN)  v = nfb[(long)node * NFD + (k - CFGF)];
    else               v = (bf16_t)0.f;
    at[j] = v;
  }
  __syncthreads();

  v8f c0 = {}, c1 = {};
#pragma unroll
  for (int kt = 0; kt < DIN_PAD / 32; ++kt) {
    v16bf a  = load_a_tile(at + kt * 32, DIN_PAD, lane);
    v16bf bA = load_b_tile(w1t, DIN_PAD, 0,  kt * 32, lane);
    v16bf bB = load_b_tile(w1t, DIN_PAD, 16, kt * 32, lane);
    c0 = WMMA_BF16(a, bA, c0);
    c1 = WMMA_BF16(a, bB, c1);
  }
  int col = lane & 15;
  int mofs = (lane >> 4) << 3;
  float b1a = b1[col], b1b = b1[16 + col];
#pragma unroll
  for (int r = 0; r < 8; ++r) {
    ht[(mofs + r) * HH + col]      = (bf16_t)lrelu(c0[r] + b1a);
    ht[(mofs + r) * HH + 16 + col] = (bf16_t)lrelu(c1[r] + b1b);
  }
  __syncthreads();

  v8f d0 = {}, d1 = {};
  {
    v16bf a  = load_a_tile(ht, HH, lane);
    v16bf bA = load_b_tile(w2t, HH, 0,  0, lane);
    v16bf bB = load_b_tile(w2t, HH, 16, 0, lane);
    d0 = WMMA_BF16(a, bA, d0);
    d1 = WMMA_BF16(a, bB, d1);
  }
  float b2a = b2[col], b2b = b2[16 + col];
#pragma unroll
  for (int r = 0; r < 8; ++r) {
    long gr = rbase + mofs + r;
    xout[gr * HH + col]      = lrelu(d0[r] + b2a);
    xout[gr * HH + 16 + col] = lrelu(d1[r] + b2b);
  }
}

// symmetric-normalized aggregation scatter: agg[d] += concat(cfg[s], x[s]) * inv[s]
__global__ void edge_scatter_kernel(const int* __restrict__ src, const int* __restrict__ dst,
                                    int E, const float* __restrict__ cfg,
                                    const float* __restrict__ x, const float* __restrict__ inv,
                                    float* __restrict__ agg) {
  int tid = blockIdx.x * blockDim.x + threadIdx.x;
  int total = (2 * E + N_NODES) * CC;
  if (tid >= total) return;
  int e = tid / CC, c = tid - e * CC;
  int s, d;
  if (e < E)            { s = src[e];      d = dst[e];      }
  else if (e < 2 * E)   { s = dst[e - E];  d = src[e - E];  }
  else                  { s = e - 2 * E;   d = s;           }
  float w = inv[s];
  const float* cs = cfg + ((long)s * CC + c) * CFGF;
  const float* xs = x   + ((long)s * CC + c) * HH;
  float* o = agg + ((long)d * CC + c) * DGC;
#pragma unroll
  for (int k = 0; k < CFGF; ++k) atomicAdd(&o[k], cs[k] * w);
#pragma unroll
  for (int k = 0; k < HH; ++k)   atomicAdd(&o[CFGF + k], xs[k] * w);
}

// x += lrelu( lrelu((agg*inv) @ W1 + b1) @ W2 + b2 )   K=50 (pad 64)
__global__ __launch_bounds__(256)
void mlp_gc_kernel(const float* __restrict__ agg, const float* __restrict__ inv,
                   const bf16_t* __restrict__ w1t, const float* __restrict__ b1,
                   const bf16_t* __restrict__ w2t, const float* __restrict__ b2,
                   float* __restrict__ xio) {
  __shared__ __align__(16) bf16_t sA[8 * 16 * DGC_PAD];
  __shared__ __align__(16) bf16_t sH[8 * 16 * HH];
  int wid = threadIdx.x >> 5, lane = threadIdx.x & 31;
  int tile = blockIdx.x * 8 + wid;
  bf16_t* at = sA + wid * 16 * DGC_PAD;
  bf16_t* ht = sH + wid * 16 * HH;
  int rbase = tile * 16;

  for (int j = lane; j < 16 * DGC_PAD; j += 32) {
    int row = j / DGC_PAD, k = j - row * DGC_PAD;
    int gr = rbase + row;
    int node = gr / CC;
    float v = (k < DGC) ? agg[(long)gr * DGC + k] * inv[node] : 0.f;
    at[j] = (bf16_t)v;
  }
  __syncthreads();

  v8f c0 = {}, c1 = {};
#pragma unroll
  for (int kt = 0; kt < DGC_PAD / 32; ++kt) {
    v16bf a  = load_a_tile(at + kt * 32, DGC_PAD, lane);
    v16bf bA = load_b_tile(w1t, DGC_PAD, 0,  kt * 32, lane);
    v16bf bB = load_b_tile(w1t, DGC_PAD, 16, kt * 32, lane);
    c0 = WMMA_BF16(a, bA, c0);
    c1 = WMMA_BF16(a, bB, c1);
  }
  int col = lane & 15;
  int mofs = (lane >> 4) << 3;
  float b1a = b1[col], b1b = b1[16 + col];
#pragma unroll
  for (int r = 0; r < 8; ++r) {
    ht[(mofs + r) * HH + col]      = (bf16_t)lrelu(c0[r] + b1a);
    ht[(mofs + r) * HH + 16 + col] = (bf16_t)lrelu(c1[r] + b1b);
  }
  __syncthreads();

  v8f d0 = {}, d1 = {};
  {
    v16bf a  = load_a_tile(ht, HH, lane);
    v16bf bA = load_b_tile(w2t, HH, 0,  0, lane);
    v16bf bB = load_b_tile(w2t, HH, 16, 0, lane);
    d0 = WMMA_BF16(a, bA, d0);
    d1 = WMMA_BF16(a, bB, d1);
  }
  float b2a = b2[col], b2b = b2[16 + col];
#pragma unroll
  for (int r = 0; r < 8; ++r) {
    long gr = rbase + mofs + r;
    float* px = xio + gr * HH;
    px[col]      += lrelu(d0[r] + b2a);
    px[16 + col] += lrelu(d1[r] + b2b);
  }
}

// ---------------- tail kernels ----------------

__global__ void select_kernel(float* __restrict__ xf, const float* __restrict__ xb,
                              const unsigned char* __restrict__ sel) {
  int tid = blockIdx.x * blockDim.x + threadIdx.x;
  if (tid >= ROWS * HH) return;
  int node = tid / (CC * HH);
  if (sel[node]) xf[tid] = xb[tid];
}

__global__ void group_reduce_kernel(const float* __restrict__ x, const int* __restrict__ op_gid,
                                    float* __restrict__ op_sum, float* __restrict__ cnt) {
  int tid = blockIdx.x * blockDim.x + threadIdx.x;
  if (tid >= N_NODES * CC) return;
  int node = tid / CC, c = tid - node * CC;
  int g = op_gid[node];
  const float* xs = x + (long)tid * HH;
  float* o = op_sum + ((long)g * CC + c) * HH;
#pragma unroll
  for (int k = 0; k < HH; ++k) atomicAdd(&o[k], xs[k]);
  if (c == 0) atomicAdd(&cnt[g], 1.f);
}

__global__ void cfg_reduce_kernel(const float* __restrict__ x, const int* __restrict__ csrc,
                                  const int* __restrict__ cdst, const int* __restrict__ cfg_gid,
                                  float* __restrict__ cfg_sum) {
  int tid = blockIdx.x * blockDim.x + threadIdx.x;
  if (tid >= NCFG * CC) return;
  int j = tid / CC, c = tid - j * CC;
  int g = cfg_gid[csrc[j]];
  int node = cdst[j];
  const float* xs = x + ((long)node * CC + c) * HH;
  float* o = cfg_sum + ((long)g * CC + c) * HH;
#pragma unroll
  for (int k = 0; k < HH; ++k) atomicAdd(&o[k], xs[k]);
}

// feats = [op_mean, l2(op_sum), l2(cfg_sum)]; out = lrelu(feats@W1)@W2 ; 80 outputs
__global__ void head_kernel(const float* __restrict__ op_sum, const float* __restrict__ cfg_sum,
                            const float* __restrict__ cnt, const float* __restrict__ pw1,
                            const float* __restrict__ pw2, float* __restrict__ out) {
  int tid = threadIdx.x;
  if (tid >= GG * CC) return;
  int g = tid / CC, c = tid - g * CC;
  const float* os = op_sum  + ((long)g * CC + c) * HH;
  const float* cs = cfg_sum + ((long)g * CC + c) * HH;
  float m = cnt[g];
  float so = 0.f, sc = 0.f;
  for (int k = 0; k < HH; ++k) { so += os[k] * os[k]; sc += cs[k] * cs[k]; }
  float io = rsqrtf(fmaxf(so, 1e-12f));
  float ic = rsqrtf(fmaxf(sc, 1e-12f));
  float f[3 * HH];
  for (int k = 0; k < HH; ++k) {
    f[k]          = os[k] / m;
    f[HH + k]     = os[k] * io;
    f[2 * HH + k] = cs[k] * ic;
  }
  float acc = 0.f;
  for (int j = 0; j < HH; ++j) {
    float h = 0.f;
    for (int k = 0; k < 3 * HH; ++k) h += f[k] * pw1[k * HH + j];
    acc += lrelu(h) * pw2[j];
  }
  out[tid] = acc;
}

// ---------------- driver ----------------
extern "C" void kernel_launch(void* const* d_in, const int* in_sizes, int n_in,
                              void* d_out, int out_size, void* d_ws, size_t ws_size,
                              hipStream_t stream) {
  (void)n_in; (void)out_size; (void)ws_size;
  const float* op_feats     = (const float*)d_in[0];
  const float* config_feats = (const float*)d_in[1];
  const int*   op_ids       = (const int*)d_in[2];
  const unsigned char* sel  = (const unsigned char*)d_in[3];  // jax bool = 1 byte
  const int*   feed_src     = (const int*)d_in[4];
  const int*   feed_dst     = (const int*)d_in[5];
  const int*   sfeed_src    = (const int*)d_in[6];
  const int*   sfeed_dst    = (const int*)d_in[7];
  const int*   config_src   = (const int*)d_in[8];
  const int*   config_dst   = (const int*)d_in[9];
  const int*   sconfig_src  = (const int*)d_in[10];
  const int*   sconfig_dst  = (const int*)d_in[11];
  const int*   op_gid       = (const int*)d_in[12];
  const int*   cfg_gid      = (const int*)d_in[13];
  const float* emb          = (const float*)d_in[14];
  const float* pre_w1       = (const float*)d_in[15];
  const float* pre_b1       = (const float*)d_in[16];
  const float* pre_w2       = (const float*)d_in[17];
  const float* pre_b2       = (const float*)d_in[18];
  const float* gc_w1[2]     = {(const float*)d_in[19], (const float*)d_in[23]};
  const float* gc_b1[2]     = {(const float*)d_in[20], (const float*)d_in[24]};
  const float* gc_w2[2]     = {(const float*)d_in[21], (const float*)d_in[25]};
  const float* gc_b2[2]     = {(const float*)d_in[22], (const float*)d_in[26]};
  const float* post_w1      = (const float*)d_in[27];
  const float* post_w2      = (const float*)d_in[28];

  int E  = in_sizes[4];
  int ES = in_sizes[6];

  size_t off = 0;
  auto alloc = [&](size_t bytes) -> void* {
    void* p = (char*)d_ws + off;
    off += (bytes + 255) & ~(size_t)255;
    return p;
  };
  bf16_t* nfb   = (bf16_t*)alloc((size_t)N_NODES * NFD * sizeof(bf16_t));
  float*  cfg   = (float*)alloc((size_t)N_NODES * CC * CFGF * sizeof(float));
  float*  deg   = (float*)alloc((size_t)N_NODES * sizeof(float));
  float*  invd  = (float*)alloc((size_t)N_NODES * sizeof(float));
  float*  xf    = (float*)alloc((size_t)ROWS * HH * sizeof(float));
  float*  xb    = (float*)alloc((size_t)ROWS * HH * sizeof(float));
  float*  agg   = (float*)alloc((size_t)ROWS * DGC * sizeof(float));
  bf16_t* wpre1 = (bf16_t*)alloc((size_t)HH * DIN_PAD * sizeof(bf16_t));
  bf16_t* wpre2 = (bf16_t*)alloc((size_t)HH * HH * sizeof(bf16_t));
  bf16_t* wgc1t[2], *wgc2t[2];
  for (int l = 0; l < 2; ++l) {
    wgc1t[l] = (bf16_t*)alloc((size_t)HH * DGC_PAD * sizeof(bf16_t));
    wgc2t[l] = (bf16_t*)alloc((size_t)HH * HH * sizeof(bf16_t));
  }
  float* op_sum  = (float*)alloc((size_t)GG * CC * HH * sizeof(float));
  float* cfg_sum = (float*)alloc((size_t)GG * CC * HH * sizeof(float));
  float* cnt     = (float*)alloc((size_t)GG * sizeof(float));

  const int TB = 256;
  // pack tiny weights into W^T bf16 (B-fragment friendly)
  pack_wT_kernel<<<(HH * DIN_PAD + TB - 1) / TB, TB, 0, stream>>>(pre_w1, wpre1, DIN, DIN_PAD);
  pack_wT_kernel<<<(HH * HH + TB - 1) / TB, TB, 0, stream>>>(pre_w2, wpre2, HH, HH);
  for (int l = 0; l < 2; ++l) {
    pack_wT_kernel<<<(HH * DGC_PAD + TB - 1) / TB, TB, 0, stream>>>(gc_w1[l], wgc1t[l], DGC, DGC_PAD);
    pack_wT_kernel<<<(HH * HH + TB - 1) / TB, TB, 0, stream>>>(gc_w2[l], wgc2t[l], HH, HH);
  }
  build_nf_kernel<<<(N_NODES * NFD + TB - 1) / TB, TB, 0, stream>>>(op_feats, emb, op_ids, nfb);

  for (int pass = 0; pass < 2; ++pass) {
    const int* src  = pass ? sfeed_src : feed_src;
    const int* dst  = pass ? sfeed_dst : feed_dst;
    int Ecur        = pass ? ES : E;
    const int* csrc = pass ? sconfig_src : config_src;
    const int* cdst = pass ? sconfig_dst : config_dst;
    float* x        = pass ? xb : xf;

    hipMemsetAsync(cfg, 0, (size_t)N_NODES * CC * CFGF * sizeof(float), stream);
    cfg_scatter_kernel<<<(NCFG * CC + TB - 1) / TB, TB, 0, stream>>>(config_feats, csrc, cdst, cfg);

    hipMemsetAsync(deg, 0, (size_t)N_NODES * sizeof(float), stream);
    deg_kernel<<<(2 * Ecur + TB - 1) / TB, TB, 0, stream>>>(src, dst, Ecur, deg);
    inv_kernel<<<(N_NODES + TB - 1) / TB, TB, 0, stream>>>(deg, invd);

    mlp_pre_kernel<<<NTILES / 8, 256, 0, stream>>>(nfb, cfg, wpre1, pre_b1, wpre2, pre_b2, x);

    for (int l = 0; l < 2; ++l) {
      hipMemsetAsync(agg, 0, (size_t)ROWS * DGC * sizeof(float), stream);
      int tot = (2 * Ecur + N_NODES) * CC;
      edge_scatter_kernel<<<(tot + TB - 1) / TB, TB, 0, stream>>>(src, dst, Ecur, cfg, x, invd, agg);
      mlp_gc_kernel<<<NTILES / 8, 256, 0, stream>>>(agg, invd, wgc1t[l], gc_b1[l], wgc2t[l], gc_b2[l], x);
    }
  }

  select_kernel<<<(ROWS * HH + TB - 1) / TB, TB, 0, stream>>>(xf, xb, sel);

  hipMemsetAsync(op_sum,  0, (size_t)GG * CC * HH * sizeof(float), stream);
  hipMemsetAsync(cfg_sum, 0, (size_t)GG * CC * HH * sizeof(float), stream);
  hipMemsetAsync(cnt,     0, (size_t)GG * sizeof(float), stream);
  group_reduce_kernel<<<(N_NODES * CC + TB - 1) / TB, TB, 0, stream>>>(xf, op_gid, op_sum, cnt);
  cfg_reduce_kernel<<<(NCFG * CC + TB - 1) / TB, TB, 0, stream>>>(xf, config_src, config_dst, cfg_gid, cfg_sum);
  head_kernel<<<1, 128, 0, stream>>>(op_sum, cfg_sum, cnt, post_w1, post_w2, (float*)d_out);
}
